// EncoderMOE_15796889714928
// MI455X (gfx1250) — matrix operand order repaired and checked by reading, compiled, not run on hardware
//
#include <hip/hip_runtime.h>

// ---- problem dims ----
#define NB    8
#define SEQ   512
#define DM    768
#define NH    12
#define DHD   64
#define NE    8
#define FF    3072
#define TTOK  4096   // NB*SEQ

typedef __attribute__((ext_vector_type(16))) __bf16 v16bf;
typedef __attribute__((ext_vector_type(8)))  float  v8f;
typedef __attribute__((ext_vector_type(4)))  unsigned int u32x4;
typedef __attribute__((ext_vector_type(4)))  int i32x4;
typedef __attribute__((ext_vector_type(8)))  int i32x8;

#if defined(__has_builtin)
#  if __has_builtin(__builtin_amdgcn_tensor_load_to_lds)
#    define HAVE_TDM 1
#  else
#    define HAVE_TDM 0
#  endif
#  if __has_builtin(__builtin_amdgcn_s_wait_tensorcnt)
#    define HAVE_WTC 1
#  else
#    define HAVE_WTC 0
#  endif
#else
#  define HAVE_TDM 0
#  define HAVE_WTC 0
#endif

// ---------- bf16 helpers (bit-exact RNE convert) ----------
__device__ inline __bf16 f2bf(float x) {
  unsigned u = __float_as_uint(x);
  unsigned r = (u + 0x7FFFu + ((u >> 16) & 1u)) >> 16;
  unsigned short s = (unsigned short)r;
  __bf16 b; __builtin_memcpy(&b, &s, 2); return b;
}
__device__ inline float bf2f(__bf16 x) {
  unsigned short s; __builtin_memcpy(&s, &x, 2);
  unsigned u = ((unsigned)s) << 16; return __uint_as_float(u);
}
__device__ inline v8f zero8() {
  v8f z = {0.f,0.f,0.f,0.f,0.f,0.f,0.f,0.f}; return z;
}

// ---------- WMMA fragment loaders ----------
// A fragment (16x32 bf16, row-major, lda elements):
//   lane<16 : row m0+lane,    K {k0..k0+7} and {k0+16..k0+23}
//   lane>=16: row m0+lane-16, K {k0+8..k0+15} and {k0+24..k0+31}
__device__ inline v16bf load_a(const __bf16* A, int lda, int m0, int k0, int lane) {
  const __bf16* p = A + (size_t)(m0 + (lane & 15)) * lda + k0 + ((lane >> 4) << 3);
  union { v16bf v; uint4 u[2]; } f;
  f.u[0] = *(const uint4*)(p);
  f.u[1] = *(const uint4*)(p + 16);
  return f.v;
}
// B fragment (32x16 bf16) from W^T stored row-major [N][K] (ldw elements):
//   lane<16 : col n0+lane,    K {k0..k0+7} and {k0+16..k0+23}
//   lane>=16: col n0+lane-16, K {k0+8..k0+15} and {k0+24..k0+31}
__device__ inline v16bf load_b(const __bf16* Wt, int ldw, int n0, int k0, int lane) {
  const __bf16* p = Wt + (size_t)(n0 + (lane & 15)) * ldw + k0 + ((lane >> 4) << 3);
  union { v16bf v; uint4 u[2]; } f;
  f.u[0] = *(const uint4*)(p);
  f.u[1] = *(const uint4*)(p + 16);
  return f.v;
}
__device__ inline v8f wmma_bf16(v16bf a, v16bf b, v8f c) {
  return __builtin_amdgcn_wmma_f32_16x16x32_bf16(false, a, false, b, (short)0, c, false, false);
}

// ---------- double-buffered 16x64 GEMM tile: acc += A[m0:,:kdim] * Wt^T ----------
__device__ inline void gemm16x64_db(const __bf16* __restrict__ A, int lda,
                                    const __bf16* __restrict__ Wt, int ldw,
                                    int m0, int n0, int kdim, int lane, v8f acc[4]) {
  v16bf a  = load_a(A, lda, m0, 0, lane);
  v16bf b0 = load_b(Wt, ldw, n0 +  0, 0, lane);
  v16bf b1 = load_b(Wt, ldw, n0 + 16, 0, lane);
  v16bf b2 = load_b(Wt, ldw, n0 + 32, 0, lane);
  v16bf b3 = load_b(Wt, ldw, n0 + 48, 0, lane);
  for (int k0 = 32; k0 < kdim; k0 += 32) {
    v16bf an = load_a(A, lda, m0, k0, lane);
    v16bf c0 = load_b(Wt, ldw, n0 +  0, k0, lane);
    v16bf c1 = load_b(Wt, ldw, n0 + 16, k0, lane);
    v16bf c2 = load_b(Wt, ldw, n0 + 32, k0, lane);
    v16bf c3 = load_b(Wt, ldw, n0 + 48, k0, lane);
    acc[0] = wmma_bf16(a, b0, acc[0]);
    acc[1] = wmma_bf16(a, b1, acc[1]);
    acc[2] = wmma_bf16(a, b2, acc[2]);
    acc[3] = wmma_bf16(a, b3, acc[3]);
    a = an; b0 = c0; b1 = c1; b2 = c2; b3 = c3;
  }
  acc[0] = wmma_bf16(a, b0, acc[0]);
  acc[1] = wmma_bf16(a, b1, acc[1]);
  acc[2] = wmma_bf16(a, b2, acc[2]);
  acc[3] = wmma_bf16(a, b3, acc[3]);
}

__device__ inline float gelu_f(float x) {
  float inner = 0.7978845608028654f * (x + 0.044715f * x * x * x);
  return 0.5f * x * (1.0f + tanhf(inner));
}

// ---------- weight convert + transpose: Wt[n*Kd + k] = W[k*Nd + n] ----------
__global__ void convt_kernel(const float* __restrict__ W, __bf16* __restrict__ Wt, int Kd, int Nd) {
  size_t per = (size_t)Kd * Nd;
  const float* Wm = W + per * blockIdx.y;
  __bf16* Wo = Wt + per * blockIdx.y;
  size_t idx = (size_t)blockIdx.x * blockDim.x + threadIdx.x;
  if (idx >= per) return;
  int n = (int)(idx / Kd);
  int k = (int)(idx % Kd);
  Wo[idx] = f2bf(Wm[(size_t)k * Nd + n]);
}

__global__ void zero_counts_kernel(int* counts) {
  if (threadIdx.x < NE) counts[threadIdx.x] = 0;
}

// ---------- LayerNorm: one wave per row, fp32 in -> bf16 out ----------
__global__ void ln_kernel(const float* __restrict__ x, const float* __restrict__ g,
                          const float* __restrict__ b, __bf16* __restrict__ out) {
  int t = blockIdx.x;
  int lane = threadIdx.x;
  const float* row = x + (size_t)t * DM;
  float s1 = 0.f, s2 = 0.f;
  for (int c = lane; c < DM; c += 32) { float v = row[c]; s1 += v; s2 += v * v; }
  for (int off = 16; off; off >>= 1) { s1 += __shfl_xor(s1, off, 32); s2 += __shfl_xor(s2, off, 32); }
  float mu = s1 * (1.f / DM);
  float var = s2 * (1.f / DM) - mu * mu;
  float rs = rsqrtf(var + 1e-5f);
  __bf16* o = out + (size_t)t * DM;
  for (int c = lane; c < DM; c += 32)
    o[c] = f2bf((row[c] - mu) * rs * g[c] + b[c]);
}

// ---------- Q/K projection: 4 waves/block, each a 16x64 tile -> [BH][S][DH] ----------
__global__ __launch_bounds__(128)
void proj_qk_kernel(const __bf16* __restrict__ A, const __bf16* __restrict__ Wt,
                    __bf16* __restrict__ out) {
  int tid = threadIdx.x, w = tid >> 5, lane = tid & 31;
  int m0 = blockIdx.x * 16, n0 = (blockIdx.y * 4 + w) * 64;
  v8f acc[4]; for (int j = 0; j < 4; ++j) acc[j] = zero8();
  gemm16x64_db(A, DM, Wt, DM, m0, n0, DM, lane, acc);
  int hi = lane >> 4, lo = lane & 15;
#pragma unroll
  for (int j = 0; j < 4; ++j)
#pragma unroll
    for (int r = 0; r < 8; ++r) {
      int col = n0 + 16 * j + lo;
      int m = m0 + r + hi * 8;
      int bI = m >> 9, s = m & 511, hI = col >> 6, dh = col & 63;
      out[((((size_t)bI * NH + hI) * SEQ) + s) * DHD + dh] = f2bf(acc[j][r]);
    }
}

// ---------- V projection, stored transposed: [BH][DH][S] ----------
__global__ __launch_bounds__(128)
void proj_v_kernel(const __bf16* __restrict__ A, const __bf16* __restrict__ Wt,
                   __bf16* __restrict__ out) {
  int tid = threadIdx.x, w = tid >> 5, lane = tid & 31;
  int m0 = blockIdx.x * 16, n0 = (blockIdx.y * 4 + w) * 64;
  v8f acc[4]; for (int j = 0; j < 4; ++j) acc[j] = zero8();
  gemm16x64_db(A, DM, Wt, DM, m0, n0, DM, lane, acc);
  int hi = lane >> 4, lo = lane & 15;
#pragma unroll
  for (int j = 0; j < 4; ++j)
#pragma unroll
    for (int r = 0; r < 8; ++r) {
      int col = n0 + 16 * j + lo;
      int m = m0 + r + hi * 8;
      int bI = m >> 9, s = m & 511, hI = col >> 6, dh = col & 63;
      out[((((size_t)bI * NH + hI) * DHD) + dh) * SEQ + s] = f2bf(acc[j][r]);
    }
}

// ---------- attention: 4 waves/block, each owns a 16-row q-tile + 16KB bf16 LDS ----------
__global__ __launch_bounds__(128)
void attn_kernel(const __bf16* __restrict__ Q, const __bf16* __restrict__ Kb,
                 const __bf16* __restrict__ Vt, __bf16* __restrict__ O) {
  __shared__ __align__(16) __bf16 scbuf[4][16 * SEQ];   // 64 KB total, 16 KB per wave
  int tid = threadIdx.x, w = tid >> 5, lane = tid & 31;
  int qt = blockIdx.x * 4 + w, bh = blockIdx.y;
  __bf16* S = scbuf[w];
  const __bf16* Qp = Q  + (size_t)bh * SEQ * DHD;
  const __bf16* Kp = Kb + (size_t)bh * SEQ * DHD;
  const __bf16* Vp = Vt + (size_t)bh * DHD * SEQ;
  int m0 = qt * 16;
  int hi = lane >> 4, lo = lane & 15;

  // phase 1: S = softmax-in(Q K^T / 8). Q fragments invariant, K double-buffered.
  v16bf aq0 = load_a(Qp, DHD, m0, 0, lane);
  v16bf aq1 = load_a(Qp, DHD, m0, 32, lane);
  v16bf b0 = load_b(Kp, DHD, 0, 0, lane);
  v16bf b1 = load_b(Kp, DHD, 0, 32, lane);
  for (int kt = 0; kt < SEQ / 16; ++kt) {
    int ktn = (kt + 1) & (SEQ / 16 - 1);
    v16bf c0 = load_b(Kp, DHD, ktn * 16, 0, lane);
    v16bf c1 = load_b(Kp, DHD, ktn * 16, 32, lane);
    v8f acc = zero8();
    acc = wmma_bf16(aq0, b0, acc);
    acc = wmma_bf16(aq1, b1, acc);
#pragma unroll
    for (int r = 0; r < 8; ++r)
      S[(r + hi * 8) * SEQ + kt * 16 + lo] = f2bf(acc[r] * 0.125f);
    b0 = c0; b1 = c1;
  }

  // per-row wave softmax (in place, bf16)
  for (int r = 0; r < 16; ++r) {
    float m = -1e30f;
    for (int c = lane; c < SEQ; c += 32) m = fmaxf(m, bf2f(S[r * SEQ + c]));
    for (int off = 16; off; off >>= 1) m = fmaxf(m, __shfl_xor(m, off, 32));
    float s = 0.f;
    for (int c = lane; c < SEQ; c += 32) {
      float e = __expf(bf2f(S[r * SEQ + c]) - m);
      S[r * SEQ + c] = f2bf(e); s += e;
    }
    for (int off = 16; off; off >>= 1) s += __shfl_xor(s, off, 32);
    float inv = 1.f / s;
    for (int c = lane; c < SEQ; c += 32)
      S[r * SEQ + c] = f2bf(bf2f(S[r * SEQ + c]) * inv);
  }

  // phase 2: O = P V (double-buffered, A from LDS, B from Vt)
  int bI = bh / NH, hI = bh % NH;
  for (int nt = 0; nt < DHD / 16; ++nt) {
    v8f acc = zero8();
    v16bf a = load_a(S, SEQ, 0, 0, lane);
    v16bf b = load_b(Vp, SEQ, nt * 16, 0, lane);
    for (int k0 = 32; k0 < SEQ; k0 += 32) {
      v16bf an = load_a(S, SEQ, 0, k0, lane);
      v16bf bn = load_b(Vp, SEQ, nt * 16, k0, lane);
      acc = wmma_bf16(a, b, acc);
      a = an; b = bn;
    }
    acc = wmma_bf16(a, b, acc);
#pragma unroll
    for (int r = 0; r < 8; ++r) {
      int m = m0 + r + hi * 8;
      int t = bI * SEQ + m;
      int col = hI * DHD + nt * 16 + lo;
      O[(size_t)t * DM + col] = f2bf(acc[r]);
    }
  }
}

// ---------- O projection + residual -> fp32 d_out ----------
__global__ __launch_bounds__(128)
void proj_o_kernel(const __bf16* __restrict__ A, const __bf16* __restrict__ Wt,
                   const float* __restrict__ xres, float* __restrict__ out) {
  int tid = threadIdx.x, w = tid >> 5, lane = tid & 31;
  int m0 = blockIdx.x * 16, n0 = (blockIdx.y * 4 + w) * 64;
  v8f acc[4]; for (int j = 0; j < 4; ++j) acc[j] = zero8();
  gemm16x64_db(A, DM, Wt, DM, m0, n0, DM, lane, acc);
  int hi = lane >> 4, lo = lane & 15;
#pragma unroll
  for (int j = 0; j < 4; ++j)
#pragma unroll
    for (int r = 0; r < 8; ++r) {
      int col = n0 + 16 * j + lo;
      int m = m0 + r + hi * 8;
      size_t idx = (size_t)m * DM + col;
      out[idx] = acc[j][r] + xres[idx];
    }
}

// ---------- router: logits, biased top-2, softmax gates, gather per expert ----------
__global__ void router_kernel(const __bf16* __restrict__ h2, const float* __restrict__ gw,
                              const float* __restrict__ gb, const float* __restrict__ be,
                              int* __restrict__ counts, int* __restrict__ etok,
                              float* __restrict__ egate) {
  int t = blockIdx.x * blockDim.x + threadIdx.x;
  if (t >= TTOK) return;
  float lg[NE];
#pragma unroll
  for (int e = 0; e < NE; ++e) lg[e] = gb[e];
  const __bf16* hr = h2 + (size_t)t * DM;
  for (int d = 0; d < DM; ++d) {
    float hv = bf2f(hr[d]);
    const float* gr = gw + (size_t)d * NE;
#pragma unroll
    for (int e = 0; e < NE; ++e) lg[e] += hv * gr[e];
  }
  int i0 = 0, i1 = 0;
  float s0 = -1e30f, s1 = -1e30f;
#pragma unroll
  for (int e = 0; e < NE; ++e) {
    float s = lg[e] + be[e];
    if (s > s0) { s1 = s0; i1 = i0; s0 = s; i0 = e; }
    else if (s > s1) { s1 = s; i1 = e; }
  }
  float l0 = lg[i0], l1 = lg[i1];
  float mx = fmaxf(l0, l1);
  float e0 = __expf(l0 - mx), e1 = __expf(l1 - mx);
  float inv = 1.f / (e0 + e1);
  int p0 = atomicAdd(&counts[i0], 1);
  etok[i0 * TTOK + p0] = t; egate[i0 * TTOK + p0] = e0 * inv;
  int p1 = atomicAdd(&counts[i1], 1);
  etok[i1 * TTOK + p1] = t; egate[i1 * TTOK + p1] = e1 * inv;
}

// ---------- fused expert FFN: 16 tokens/WG, mid (16x3072 bf16) lives in LDS ----------
__global__ __launch_bounds__(256)
void ffn_kernel(const __bf16* __restrict__ h2, const __bf16* __restrict__ w1t,
                const float* __restrict__ b1, const __bf16* __restrict__ w2t,
                const float* __restrict__ b2, const int* __restrict__ counts,
                const int* __restrict__ etok, const float* __restrict__ egate,
                float* __restrict__ out) {
  extern __shared__ __align__(16) char smem[];
  __bf16* hT  = (__bf16*)smem;          // 16*768  bf16 = 24 KB
  __bf16* mid = hT + 16 * DM;           // 16*3072 bf16 = 96 KB
  __shared__ int   toks[16];
  __shared__ float gts[16];

  int e = blockIdx.y;
  int tile = blockIdx.x;
  int cnt = counts[e];
  if (tile * 16 >= cnt) return;
  int tid = threadIdx.x;

  if (tid < 16) {
    int row = tile * 16 + tid;
    if (row < cnt) { toks[tid] = etok[e * TTOK + row]; gts[tid] = egate[e * TTOK + row]; }
    else           { toks[tid] = 0;                    gts[tid] = 0.f; }
  }
  __syncthreads();

#if HAVE_TDM
  // Tensor Data Mover gather-mode load: 16 scattered token rows (16-bit row
  // indices = token ids) of h2 -> contiguous 16x768 bf16 LDS tile.
  if (tid < 32) {
    unsigned long long ga = (unsigned long long)(uintptr_t)h2;
    unsigned ldsa = (unsigned)(uintptr_t)hT;
    u32x4 g0;
    g0[0] = 0x80000001u;                                     // count=1, gather_mode=1, 16b idx
    g0[1] = ldsa;                                            // lds_addr
    g0[2] = (unsigned)(ga & 0xFFFFFFFFu);                    // global_addr lo
    g0[3] = (unsigned)((ga >> 32) & 0x01FFFFFFu) | (2u << 30); // global_addr hi | type=2
    i32x8 g1;
    g1[0] = (int)(1u << 16);                                 // data_size = 2 bytes
    g1[1] = (int)(((unsigned)DM & 0xFFFFu) << 16);           // tensor_dim0 lo16
    g1[2] = (int)((((unsigned)DM >> 16) & 0xFFFFu) | (((unsigned)TTOK & 0xFFFFu) << 16));
    g1[3] = (int)((((unsigned)TTOK >> 16) & 0xFFFFu) | (((unsigned)DM & 0xFFFFu) << 16)); // tile_dim0
    g1[4] = 16;                                              // tile_dim1 = #valid indices
    g1[5] = DM;                                              // tensor_dim0_stride lo32
    g1[6] = 0;
    g1[7] = 0;
    int iw[8];
#pragma unroll
    for (int j = 0; j < 8; ++j)
      iw[j] = __builtin_amdgcn_readfirstlane(toks[2 * j] | (toks[2 * j + 1] << 16));
    i32x4 g2; g2[0] = iw[0]; g2[1] = iw[1]; g2[2] = iw[2]; g2[3] = iw[3];
    i32x4 g3; g3[0] = iw[4]; g3[1] = iw[5]; g3[2] = iw[6]; g3[3] = iw[7];
#if __clang_major__ >= 23
    i32x8 g4 = {0,0,0,0,0,0,0,0};
    __builtin_amdgcn_tensor_load_to_lds(g0, g1, g2, g3, g4, 0);
#else
    __builtin_amdgcn_tensor_load_to_lds(g0, g1, g2, g3, 0);
#endif
#if HAVE_WTC
    __builtin_amdgcn_s_wait_tensorcnt(0);
#else
    asm volatile("s_wait_tensorcnt 0x0" ::: "memory");
#endif
  }
  __syncthreads();
#else
  for (int idx = tid; idx < 16 * DM; idx += 256)
    hT[idx] = h2[(size_t)toks[idx / DM] * DM + (idx % DM)];
  __syncthreads();
#endif

  int w = tid >> 5, lane = tid & 31, hi = lane >> 4, lo = lane & 15;
  const __bf16* W1 = w1t + (size_t)e * FF * DM;   // [F][D]
  const float*  B1 = b1 + e * FF;

  // pass 1: mid = gelu(h @ w1 + b1); each wave covers 384 F-columns (6 groups of 64)
  for (int g = 0; g < 6; ++g) {
    int n0 = w * 384 + g * 64;
    if (g < 5)  // prefetch next group's weight rows into cache
      __builtin_prefetch(W1 + (size_t)(n0 + 64 + lo) * DM, 0, 0);
    v8f acc[4]; for (int j = 0; j < 4; ++j) acc[j] = zero8();
    gemm16x64_db(hT, DM, W1, DM, 0, n0, DM, lane, acc);
#pragma unroll
    for (int j = 0; j < 4; ++j)
#pragma unroll
      for (int r = 0; r < 8; ++r) {
        int col = n0 + 16 * j + lo;
        int m = r + hi * 8;
        mid[m * FF + col] = f2bf(gelu_f(acc[j][r] + B1[col]));
      }
  }
  __syncthreads();

  const __bf16* W2 = w2t + (size_t)e * DM * FF;   // [D][F]
  const float*  B2 = b2 + e * DM;

  // pass 2: y = gate * (mid @ w2 + b2); each wave covers 96 D-columns (6 tiles of 16)
  for (int nt = 0; nt < 6; ++nt) {
    int n0 = w * 96 + nt * 16;
    if (nt < 5)
      __builtin_prefetch(W2 + (size_t)(n0 + 16 + lo) * FF, 0, 0);
    v8f acc = zero8();
    v16bf a = load_a(mid, FF, 0, 0, lane);
    v16bf b = load_b(W2, FF, n0, 0, lane);
    for (int k0 = 32; k0 < FF; k0 += 32) {
      v16bf an = load_a(mid, FF, 0, k0, lane);
      v16bf bn = load_b(W2, FF, n0, k0, lane);
      acc = wmma_bf16(a, b, acc);
      a = an; b = bn;
    }
    acc = wmma_bf16(a, b, acc);
#pragma unroll
    for (int r = 0; r < 8; ++r) {
      int m = r + hi * 8;
      float gte = gts[m];
      if (gte != 0.f) {
        int col = n0 + lo;
        float v = gte * (acc[r] + B2[col]);
        atomicAdd(&out[(size_t)toks[m] * DM + col], v);
      }
    }
  }
}

// =====================================================================================
extern "C" void kernel_launch(void* const* d_in, const int* in_sizes, int n_in,
                              void* d_out, int out_size, void* d_ws, size_t ws_size,
                              hipStream_t stream) {
  (void)in_sizes; (void)n_in; (void)out_size; (void)ws_size;
  const float* x      = (const float*)d_in[0];
  const float* wq     = (const float*)d_in[1];
  const float* wk     = (const float*)d_in[2];
  const float* wv     = (const float*)d_in[3];
  const float* wo     = (const float*)d_in[4];
  const float* ln1_g  = (const float*)d_in[5];
  const float* ln1_b  = (const float*)d_in[6];
  const float* ln2_g  = (const float*)d_in[7];
  const float* ln2_b  = (const float*)d_in[8];
  const float* gate_w = (const float*)d_in[9];
  const float* gate_b = (const float*)d_in[10];
  const float* bias_e = (const float*)d_in[11];
  const float* w1     = (const float*)d_in[12];
  const float* b1     = (const float*)d_in[13];
  const float* w2     = (const float*)d_in[14];
  const float* b2     = (const float*)d_in[15];
  float* out = (float*)d_out;

  // workspace carve-up (256B aligned)
  char* ws = (char*)d_ws;
  size_t off = 0;
  auto alloc = [&](size_t bytes) -> char* {
    char* p = ws + off;
    off += (bytes + 255) & ~(size_t)255;
    return p;
  };
  __bf16* wq_t = (__bf16*)alloc((size_t)DM * DM * 2);
  __bf16* wk_t = (__bf16*)alloc((size_t)DM * DM * 2);
  __bf16* wv_t = (__bf16*)alloc((size_t)DM * DM * 2);
  __bf16* wo_t = (__bf16*)alloc((size_t)DM * DM * 2);
  __bf16* w1t  = (__bf16*)alloc((size_t)NE * FF * DM * 2);
  __bf16* w2t  = (__bf16*)alloc((size_t)NE * DM * FF * 2);
  __bf16* h1   = (__bf16*)alloc((size_t)TTOK * DM * 2);
  __bf16* qb   = (__bf16*)alloc((size_t)TTOK * DM * 2);
  __bf16* kb   = (__bf16*)alloc((size_t)TTOK * DM * 2);
  __bf16* vbT  = (__bf16*)alloc((size_t)TTOK * DM * 2);
  __bf16* ob   = (__bf16*)alloc((size_t)TTOK * DM * 2);
  __bf16* h2   = (__bf16*)alloc((size_t)TTOK * DM * 2);
  int*    counts = (int*)alloc(NE * sizeof(int));
  int*    etok   = (int*)alloc((size_t)NE * TTOK * sizeof(int));
  float*  egate  = (float*)alloc((size_t)NE * TTOK * sizeof(float));

  // 1) weight conversion (fp32 -> transposed bf16)
  {
    int blocks = (DM * DM + 255) / 256;
    convt_kernel<<<dim3(blocks, 1), 256, 0, stream>>>(wq, wq_t, DM, DM);
    convt_kernel<<<dim3(blocks, 1), 256, 0, stream>>>(wk, wk_t, DM, DM);
    convt_kernel<<<dim3(blocks, 1), 256, 0, stream>>>(wv, wv_t, DM, DM);
    convt_kernel<<<dim3(blocks, 1), 256, 0, stream>>>(wo, wo_t, DM, DM);
    int blocks_e = (DM * FF + 255) / 256;
    convt_kernel<<<dim3(blocks_e, NE), 256, 0, stream>>>(w1, w1t, DM, FF); // w1t[f][d]
    convt_kernel<<<dim3(blocks_e, NE), 256, 0, stream>>>(w2, w2t, FF, DM); // w2t[d][f]
  }
  zero_counts_kernel<<<1, 32, 0, stream>>>(counts);

  // 2) LN1
  ln_kernel<<<TTOK, 32, 0, stream>>>(x, ln1_g, ln1_b, h1);

  // 3) Q/K/V projections (WMMA, 4 waves/block)
  proj_qk_kernel<<<dim3(TTOK / 16, DM / 256), 128, 0, stream>>>(h1, wq_t, qb);
  proj_qk_kernel<<<dim3(TTOK / 16, DM / 256), 128, 0, stream>>>(h1, wk_t, kb);
  proj_v_kernel <<<dim3(TTOK / 16, DM / 256), 128, 0, stream>>>(h1, wv_t, vbT);

  // 4) attention (WMMA scores + in-place bf16 softmax + WMMA PV)
  attn_kernel<<<dim3(SEQ / 64, NB * NH), 128, 0, stream>>>(qb, kb, vbT, ob);

  // 5) output projection + residual -> fp32 d_out
  proj_o_kernel<<<dim3(TTOK / 16, DM / 256), 128, 0, stream>>>(ob, wo_t, x, out);

  // 6) LN2 + router
  ln_kernel<<<TTOK, 32, 0, stream>>>(out, ln2_g, ln2_b, h2);
  router_kernel<<<TTOK / 256, 256, 0, stream>>>(h2, gate_w, gate_b, bias_e,
                                                counts, etok, egate);

  // 7) fused routed expert FFN (TDM gather of token rows, WMMA x2, gelu in LDS)
  size_t smem = (size_t)16 * DM * 2 + (size_t)16 * FF * 2; // 120 KB dynamic LDS
  ffn_kernel<<<dim3(TTOK / 16, NE), 256, smem, stream>>>(h2, w1t, b1, w2t, b2,
                                                         counts, etok, egate, out);
}
